// GraphAwareCausalAttention_738734375182
// MI455X (gfx1250) — matrix operand order repaired and checked by reading, not compiled, not run on hardware
//
#include <hip/hip_runtime.h>
#include <cstdint>
#include <cstddef>

// ---------------------------------------------------------------------------
// Types for CDNA5 WMMA (wave32)
// ---------------------------------------------------------------------------
typedef __attribute__((ext_vector_type(16))) __bf16 v16bf;
typedef __attribute__((ext_vector_type(8)))  __bf16 v8bf;
typedef __attribute__((ext_vector_type(8)))  float  v8f;

#define B_SZ   2
#define T_SEQ  2048
#define D_MOD  1024
#define NH     16
#define HD     64
#define D3     3072
#define NE_CT  17

__device__ __forceinline__ v8f zero8() {
  v8f z;
#pragma unroll
  for (int i = 0; i < 8; ++i) z[i] = 0.0f;
  return z;
}

// A-matrix fragment, 16x32 bf16 (ISA 7.12.2): lane l holds row (l&15);
// VGPR 0..3 = K(kh*8 + 0..7), VGPR 4..7 = K(16 + kh*8 + 0..7), kh = l>>4.
__device__ __forceinline__ v16bf load_frag_a(const __bf16* base, size_t stride, int lane) {
  const int row = lane & 15, kh = lane >> 4;
  const __bf16* p = base + (size_t)row * stride;
  union { v16bf v; v8bf h[2]; } u;
  u.h[0] = *(const v8bf*)(p + kh * 8);
  u.h[1] = *(const v8bf*)(p + 16 + kh * 8);
  return u.v;
}

// B-matrix fragment, 32x16 bf16: lane l holds col n=(l&15);
// VGPR 0..3 = K(kh*8 + 0..7), VGPR 4..7 = K(16 + kh*8 + 0..7), kh = l>>4 (matches A).
// Requires source laid out [n][k] (k contiguous).
__device__ __forceinline__ v16bf load_frag_b(const __bf16* base, size_t stride, int lane) {
  const int n = lane & 15, kh = lane >> 4;
  const __bf16* p = base + (size_t)n * stride;
  union { v16bf v; v8bf h[2]; } u;
  u.h[0] = *(const v8bf*)(p + kh * 8);
  u.h[1] = *(const v8bf*)(p + 16 + kh * 8);
  return u.v;
}

// ---------------------------------------------------------------------------
// CDNA5 async copy: global -> LDS, 16B per lane, tracked by ASYNCcnt.
// LDS aperture keeps the wave-relative LDS offset in addr[31:0] (ISA 10.2),
// so truncating the flat pointer yields the DS address for VDST.
// ---------------------------------------------------------------------------
__device__ __forceinline__ void async_cp_b128(uint32_t lds_off, uint64_t gaddr) {
  asm volatile("global_load_async_to_lds_b128 %0, %1, off"
               :: "v"(lds_off), "v"(gaddr) : "memory");
}
__device__ __forceinline__ void wait_async0() {
  asm volatile("s_wait_asynccnt 0x0" ::: "memory");
}

// ---------------------------------------------------------------------------
// GEMM: C(MxN) = A(MxK,bf16 row-major) * Bt(NxK,bf16 row-major, i.e. B^T),
// f32 accumulate via v_wmma_f32_16x16x32_bf16.
// Block 256 thr = 8 waves; block tile 128x128, K-step 32, double-buffered
// LDS staged entirely with global_load_async_to_lds_b128.
// ---------------------------------------------------------------------------
#define BM 128
#define BN 128
#define BK 32
#define LSTR 40   // bf16 row stride: 80 B (16B-aligned, conflict-free frag reads)

// Stage a 128-row x 32-col bf16 tile (row stride gstride) into LDS [row][k].
// 128 rows * 64 B = 512 b128 chunks -> 2 async instructions per thread.
__device__ __forceinline__ void stage_tile_async(const __bf16* gsrc, int gstride,
                                                 __bf16* lbase, int tid) {
#pragma unroll
  for (int c = tid; c < 512; c += 256) {
    const int row = c >> 2, ch = c & 3;
    const uint64_t g = (uint64_t)(uintptr_t)(gsrc + (size_t)row * gstride + ch * 8);
    const uint32_t l = (uint32_t)(uintptr_t)(lbase + row * LSTR + ch * 8);
    async_cp_b128(l, g);
  }
}

template <bool BF16_OUT>
__global__ __launch_bounds__(256) void gemm_bf16wmma(
    const __bf16* __restrict__ A, const __bf16* __restrict__ Bt,
    void* __restrict__ Cout, int M, int N, int K) {
  __shared__ __align__(16) __bf16 sA[2][BM * LSTR];   // [row][k]
  __shared__ __align__(16) __bf16 sB[2][BN * LSTR];   // [n][k]

  const int tid  = threadIdx.x;
  const int lane = tid & 31;
  const int wave = tid >> 5;
  const int wm   = wave >> 1;   // 0..3
  const int wn   = wave & 1;    // 0..1
  const int row0 = blockIdx.y * BM;
  const int col0 = blockIdx.x * BN;

  v8f acc[2][4];
#pragma unroll
  for (int i = 0; i < 2; ++i)
#pragma unroll
    for (int j = 0; j < 4; ++j) acc[i][j] = zero8();

  const __bf16* Abase = A  + (size_t)row0 * K;
  const __bf16* Bbase = Bt + (size_t)col0 * K;

  // Prologue: async-fill buffer 0.
  stage_tile_async(Abase, K, &sA[0][0], tid);
  stage_tile_async(Bbase, K, &sB[0][0], tid);

  int cur = 0;
  for (int k0 = 0; k0 < K; k0 += BK) {
    wait_async0();        // current buffer's async writes landed in LDS
    __syncthreads();      // visible block-wide; prev compute done -> safe to
                          // overwrite the other buffer
    if (k0 + BK < K) {    // overlap next tile's DMA with this tile's WMMAs
      stage_tile_async(Abase + k0 + BK, K, &sA[cur ^ 1][0], tid);
      stage_tile_async(Bbase + k0 + BK, K, &sB[cur ^ 1][0], tid);
    }

    v16bf af[2], bfr[4];
#pragma unroll
    for (int mi = 0; mi < 2; ++mi)
      af[mi] = load_frag_a(&sA[cur][(wm * 32 + mi * 16) * LSTR], LSTR, lane);
#pragma unroll
    for (int ni = 0; ni < 4; ++ni)
      bfr[ni] = load_frag_b(&sB[cur][(wn * 64 + ni * 16) * LSTR], LSTR, lane);
#pragma unroll
    for (int mi = 0; mi < 2; ++mi)
#pragma unroll
      for (int ni = 0; ni < 4; ++ni)
        acc[mi][ni] = __builtin_amdgcn_wmma_f32_16x16x32_bf16(
            false, af[mi], false, bfr[ni], (short)0, acc[mi][ni], false, false);
    cur ^= 1;
  }

  // C/D layout: lane -> n=(l&15); VGPR r -> row = r + 8*(l>>4)
  const int half = lane >> 4, nlo = lane & 15;
#pragma unroll
  for (int mi = 0; mi < 2; ++mi)
#pragma unroll
    for (int ni = 0; ni < 4; ++ni) {
      const int col = col0 + wn * 64 + ni * 16 + nlo;
#pragma unroll
      for (int r = 0; r < 8; ++r) {
        const int row = row0 + wm * 32 + mi * 16 + r + 8 * half;
        const float v = acc[mi][ni][r];
        if constexpr (BF16_OUT)
          ((__bf16*)Cout)[(size_t)row * N + col] = (__bf16)v;
        else
          ((float*)Cout)[(size_t)row * N + col] = v;
      }
    }
}

// ---------------------------------------------------------------------------
// Prepass 1: f32 -> bf16 flat convert (x).
// ---------------------------------------------------------------------------
__global__ __launch_bounds__(256) void convert_bf16_kernel(
    const float* __restrict__ src, __bf16* __restrict__ dst, size_t n) {
  const size_t i = ((size_t)blockIdx.x * 256 + threadIdx.x) * 4;
  if (i + 3 < n) {
    const float4 f = *(const float4*)(src + i);
    dst[i + 0] = (__bf16)f.x; dst[i + 1] = (__bf16)f.y;
    dst[i + 2] = (__bf16)f.z; dst[i + 3] = (__bf16)f.w;
  }
}

// ---------------------------------------------------------------------------
// Prepass 2: W (KxN f32) -> Wt (NxK bf16), LDS-tiled 32x32 transpose.
// ---------------------------------------------------------------------------
__global__ __launch_bounds__(256) void transpose_conv_kernel(
    const float* __restrict__ W, __bf16* __restrict__ Wt, int K, int N) {
  __shared__ __bf16 tile[32][33];
  const int k0 = blockIdx.y * 32, n0 = blockIdx.x * 32;
  const int tr = threadIdx.x >> 3;          // 0..31
  const int tc = (threadIdx.x & 7) * 4;     // 0,4,...,28
  const float4 f = *(const float4*)(W + (size_t)(k0 + tr) * N + n0 + tc);
  tile[tr][tc + 0] = (__bf16)f.x; tile[tr][tc + 1] = (__bf16)f.y;
  tile[tr][tc + 2] = (__bf16)f.z; tile[tr][tc + 3] = (__bf16)f.w;
  __syncthreads();
#pragma unroll
  for (int i = 0; i < 4; ++i)
    Wt[(size_t)(n0 + tr) * K + k0 + tc + i] = tile[tc + i][tr];
}

// ---------------------------------------------------------------------------
// V transpose: Vt[(b*H+h)*HD+hd][t] = qkv[b*T+t][2048 + h*64 + hd]
// ---------------------------------------------------------------------------
__global__ __launch_bounds__(256) void vtrans_kernel(
    const __bf16* __restrict__ qkv, __bf16* __restrict__ Vt) {
  const int bh = blockIdx.x;            // b*1024 + h*64 + hd
  const int hd = bh & 63;
  const int h  = (bh >> 6) & 15;
  const int b  = bh >> 10;
  const __bf16* src = qkv + (size_t)(b * T_SEQ) * D3 + 2 * D_MOD + h * HD + hd;
  __bf16* dst = Vt + (size_t)bh * T_SEQ;
  for (int t = threadIdx.x; t < T_SEQ; t += 256)
    dst[t] = src[(size_t)t * D3];
}

// ---------------------------------------------------------------------------
// Flash-style causal attention with graph/edge biases.
// 1 wave per (b, h, 16-row q tile). 256 thr/block = 8 independent waves.
// ---------------------------------------------------------------------------
#define PST 40  // bf16 row stride for P staging (80 B, conflict-free)

__global__ __launch_bounds__(256) void attn_kernel(
    const __bf16* __restrict__ qkv, const __bf16* __restrict__ Vt,
    const float* __restrict__ graph_adj, const int* __restrict__ edge_types,
    const float* __restrict__ adj_bias, const float* __restrict__ edge_table,
    __bf16* __restrict__ attn_out) {
  __shared__ float s_etab[NE_CT * NH];
  __shared__ __align__(16) __bf16 sP[8 * 16 * PST];

  const int tid = threadIdx.x;
  for (int i = tid; i < NE_CT * NH; i += 256) s_etab[i] = edge_table[i];
  __syncthreads();

  const int lane = tid & 31;
  const int wave = tid >> 5;
  const int task = blockIdx.x * 8 + wave;   // B*H*(T/16) = 4096 tasks
  const int qt = task & 127;
  const int h  = (task >> 7) & 15;
  const int b  = task >> 11;
  const int q0 = qt * 16;

  const int nlo = lane & 15, half = lane >> 4;
  const float scale = 0.125f;               // 1/sqrt(64)
  const float adjb  = adj_bias[h];

  // Q fragments: 16x64 bf16 -> two 16x32 A-frags, loaded once from global.
  const __bf16* qbase = qkv + (size_t)(b * T_SEQ + q0) * D3 + h * HD;
  v16bf qa[2];
#pragma unroll
  for (int kc = 0; kc < 2; ++kc)
    qa[kc] = load_frag_a(qbase + kc * 32, D3, lane);

  v8f o[4];
#pragma unroll
  for (int f = 0; f < 4; ++f) o[f] = zero8();
  float mrow[8], lrow[8];
#pragma unroll
  for (int r = 0; r < 8; ++r) { mrow[r] = -__builtin_inff(); lrow[r] = 0.0f; }

  __bf16* myP = sP + wave * 16 * PST;
  const __bf16* kbase  = qkv + (size_t)(b * T_SEQ) * D3 + D_MOD + h * HD;
  const __bf16* vtbase = Vt + (size_t)(b * NH + h) * HD * T_SEQ;

  for (int j0 = 0; j0 <= q0 + 15; j0 += 32) {
    // ---- S = Q*K^T (f32 accum), 32 keys per iteration ----
    v8f s[2];
    s[0] = zero8(); s[1] = zero8();
#pragma unroll
    for (int sf = 0; sf < 2; ++sf) {
      const int jb = j0 + sf * 16;
#pragma unroll
      for (int kc = 0; kc < 2; ++kc) {
        v16bf kb = load_frag_b(kbase + (size_t)jb * D3 + kc * 32, D3, lane);
        s[sf] = __builtin_amdgcn_wmma_f32_16x16x32_bf16(
            false, qa[kc], false, kb, (short)0, s[sf], false, false);
      }
    }

    // Prefetch next tile's bias rows into GL2 (global_prefetch_b8).
    if (j0 + 32 <= q0 + 15) {
      const size_t nidx =
          (size_t)(b * T_SEQ + q0 + 8 * half) * T_SEQ + j0 + 32 + nlo;
      __builtin_prefetch(graph_adj + nidx, 0, 1);
      __builtin_prefetch(edge_types + nidx, 0, 1);
    }

    // ---- bias gather + causal mask (memory-bound part, L2-resident) ----
#pragma unroll
    for (int sf = 0; sf < 2; ++sf) {
      const int key = j0 + sf * 16 + nlo;
#pragma unroll
      for (int r = 0; r < 8; ++r) {
        const int qrow = q0 + r + 8 * half;
        const size_t idx = (size_t)(b * T_SEQ + qrow) * T_SEQ + key;
        float v = s[sf][r] * scale + adjb * graph_adj[idx] +
                  s_etab[edge_types[idx] * NH + h];
        if (key > qrow) v = -__builtin_inff();
        s[sf][r] = v;
      }
    }

    // ---- online softmax (row stats replicated across the 16-lane half) ----
    float mnew[8], alpha[8];
#pragma unroll
    for (int r = 0; r < 8; ++r) {
      float v = fmaxf(s[0][r], s[1][r]);
#pragma unroll
      for (int off = 1; off < 16; off <<= 1)
        v = fmaxf(v, __shfl_xor(v, off, 32));
      mnew[r]  = fmaxf(mrow[r], v);
      alpha[r] = __expf(mrow[r] - mnew[r]);
      mrow[r]  = mnew[r];
    }
#pragma unroll
    for (int sf = 0; sf < 2; ++sf)
#pragma unroll
      for (int r = 0; r < 8; ++r)
        s[sf][r] = __expf(s[sf][r] - mnew[r]);
#pragma unroll
    for (int r = 0; r < 8; ++r) {
      float rs = s[0][r] + s[1][r];
#pragma unroll
      for (int off = 1; off < 16; off <<= 1) rs += __shfl_xor(rs, off, 32);
      lrow[r] = lrow[r] * alpha[r] + rs;
    }
#pragma unroll
    for (int f = 0; f < 4; ++f)
#pragma unroll
      for (int r = 0; r < 8; ++r) o[f][r] *= alpha[r];

    // ---- P: accumulator layout -> LDS -> A fragment (16x32 bf16) ----
#pragma unroll
    for (int sf = 0; sf < 2; ++sf)
#pragma unroll
      for (int r = 0; r < 8; ++r)
        myP[(r + 8 * half) * PST + sf * 16 + nlo] = (__bf16)s[sf][r];
    // Cross-lane dependency through LDS within the wave: LDS pipe is
    // in-order per wave; drain DScnt + stop compiler reordering.
    asm volatile("s_wait_dscnt 0x0" ::: "memory");
    v16bf pa = load_frag_a(myP, PST, lane);

    // ---- O += P * V  (V^T is [hd][t] -> direct b128 B-frags) ----
#pragma unroll
    for (int f = 0; f < 4; ++f) {
      v16bf vb = load_frag_b(vtbase + (size_t)(f * 16) * T_SEQ + j0, T_SEQ, lane);
      o[f] = __builtin_amdgcn_wmma_f32_16x16x32_bf16(
          false, pa, false, vb, (short)0, o[f], false, false);
    }
  }

  // ---- normalize + store (bf16, feeds projection GEMM) ----
  float inv[8];
#pragma unroll
  for (int r = 0; r < 8; ++r) inv[r] = 1.0f / lrow[r];
#pragma unroll
  for (int f = 0; f < 4; ++f) {
    const int hd = f * 16 + nlo;
#pragma unroll
    for (int r = 0; r < 8; ++r) {
      const int qrow = q0 + r + 8 * half;
      attn_out[(size_t)(b * T_SEQ + qrow) * D_MOD + h * HD + hd] =
          (__bf16)(o[f][r] * inv[r]);
    }
  }
}

// ---------------------------------------------------------------------------
// Launch: convert/transposes -> qkv GEMM -> V transpose -> attention -> proj
// ---------------------------------------------------------------------------
extern "C" void kernel_launch(void* const* d_in, const int* in_sizes, int n_in,
                              void* d_out, int out_size, void* d_ws, size_t ws_size,
                              hipStream_t stream) {
  const float* x          = (const float*)d_in[0];
  const float* graph_adj  = (const float*)d_in[1];
  const int*   edge_types = (const int*)d_in[2];
  const float* Wqkv       = (const float*)d_in[3];
  const float* Wproj      = (const float*)d_in[4];
  const float* adj_bias   = (const float*)d_in[5];
  const float* edge_table = (const float*)d_in[6];
  float* out = (float*)d_out;

  const int M = B_SZ * T_SEQ;  // 4096
  char* ws = (char*)d_ws;
  size_t off = 0;
  __bf16* xb = (__bf16*)(ws + off);       off += (size_t)M * D_MOD * 2;       // 8 MB
  __bf16* Wqkvt = (__bf16*)(ws + off);    off += (size_t)D3 * D_MOD * 2;      // 6 MB
  __bf16* Wprojt = (__bf16*)(ws + off);   off += (size_t)D_MOD * D_MOD * 2;   // 2 MB
  __bf16* qkv = (__bf16*)(ws + off);      off += (size_t)M * D3 * 2;          // 24 MB
  __bf16* Vt = (__bf16*)(ws + off);       off += (size_t)M * D_MOD * 2;       // 8 MB
  __bf16* aout = (__bf16*)(ws + off);                                         // 8 MB

  // bf16 prepasses (one-time; unlocks async b128 LDS staging in the GEMMs)
  convert_bf16_kernel<<<(M * D_MOD) / 1024, 256, 0, stream>>>(
      x, xb, (size_t)M * D_MOD);
  transpose_conv_kernel<<<dim3(D3 / 32, D_MOD / 32), 256, 0, stream>>>(
      Wqkv, Wqkvt, D_MOD, D3);
  transpose_conv_kernel<<<dim3(D_MOD / 32, D_MOD / 32), 256, 0, stream>>>(
      Wproj, Wprojt, D_MOD, D_MOD);

  gemm_bf16wmma<true><<<dim3(D3 / BN, M / BM), 256, 0, stream>>>(
      xb, Wqkvt, qkv, M, D3, D_MOD);
  vtrans_kernel<<<B_SZ * NH * HD, 256, 0, stream>>>(qkv, Vt);
  attn_kernel<<<(B_SZ * NH * (T_SEQ / 16)) / 8, 256, 0, stream>>>(
      qkv, Vt, graph_adj, edge_types, adj_bias, edge_table, aout);
  gemm_bf16wmma<false><<<dim3(D_MOD / BN, M / BM), 256, 0, stream>>>(
      aout, Wprojt, out, M, D_MOD, D_MOD);
}
